// GATLayer_78426102825166
// MI455X (gfx1250) — matrix operand order repaired and checked by reading, not compiled, not run on hardware
//
#include <hip/hip_runtime.h>
#include <hip/hip_bf16.h>
#include <math.h>

// Problem constants (from reference): B=8, N=2048, FIN=FOUT=128, H=2
constexpr int BN = 8;
constexpr int NN = 2048;
constexpr int FN = 128;
constexpr int HN = 2;

typedef __attribute__((ext_vector_type(16))) __bf16 v16bf;
typedef __attribute__((ext_vector_type(8)))  float  v8f;
typedef __attribute__((ext_vector_type(4)))  int    v4i;

__device__ __forceinline__ __bf16 tobf(float f) { return (__bf16)f; }

// CDNA5 async global->LDS path (ASYNCcnt), guarded so the fallback keeps the
// build green if the toolchain lacks the builtins.
#if defined(__has_builtin)
#if __has_builtin(__builtin_amdgcn_global_load_async_to_lds_b128) && \
    __has_builtin(__builtin_amdgcn_s_wait_asynccnt)
#define USE_ASYNC_LDS 1
#endif
#endif

#ifdef USE_ASYNC_LDS
// Param types per hipcc diagnostic: int4 in AS1 (global) / AS3 (LDS)
typedef __attribute__((address_space(1))) v4i* gv4i_p;
typedef __attribute__((address_space(3))) v4i* lv4i_p;
__device__ __forceinline__ void async_copy32(const __bf16* src, __bf16* dst) {
    // 32 bytes = 2 x global_load_async_to_lds_b128
    __builtin_amdgcn_global_load_async_to_lds_b128((gv4i_p)src,       (lv4i_p)dst,       0, 0);
    __builtin_amdgcn_global_load_async_to_lds_b128((gv4i_p)(src + 8), (lv4i_p)(dst + 8), 0, 0);
}
#endif

// ---------------------------------------------------------------------------
// Kernel 1: Wh = h @ W (per batch, per head), bf16 WMMA with f32 accumulate.
// W is converted to bf16 once per block into LDS (K-contiguous layout ->
// B-fragments are two ds_load_b128). A-fragments come from contiguous
// float4 loads per lane (four per K-step: one 16-float run per lane
// half). Stores Wh transposed as bf16 and fuses the s_i/s_j dot products.
// grid = (B*H, N/128), block = 256 (8 waves, 16 rows per wave)
// ---------------------------------------------------------------------------
__global__ __launch_bounds__(256) void wh_kernel(
    const float* __restrict__ h, const float* __restrict__ W,
    const float* __restrict__ a, __bf16* __restrict__ Whbt,
    float* __restrict__ s_i, float* __restrict__ s_j)
{
    __shared__ __bf16 Wlds[FN][FN];   // [col][k], 32 KB

    const int bh   = blockIdx.x;
    const int b    = bh / HN;
    const int head = bh % HN;
    const int wave = threadIdx.x >> 5;
    const int lid  = threadIdx.x & 31;
    const int l16  = lid & 15;
    const int hf   = lid >> 4;
    const int rowbase = blockIdx.y * 128 + wave * 16;

    const float* hb = h + (size_t)b * NN * FN;
    const float* Wb = W + (size_t)head * FN * FN;

    // Cooperative convert W (row-major [k][c] f32) -> Wlds[c][k] bf16
    for (int idx = threadIdx.x; idx < FN * FN; idx += 256) {
        const int k = idx >> 7;
        const int c = idx & 127;
        Wlds[c][k] = tobf(Wb[idx]);
    }
    __syncthreads();

    v8f acc[8] = {};
    const int arow = rowbase + l16;

    for (int kb = 0; kb < FN; kb += 32) {
        // A fragment: 16x32 bf16, contiguous K = one 16-run per lane half
        const float* hp = hb + (size_t)arow * FN + kb;
        const float4 x0 = *(const float4*)(hp + 16 * hf);
        const float4 x1 = *(const float4*)(hp + 16 * hf + 4);
        const float4 x2 = *(const float4*)(hp + 16 * hf + 8);
        const float4 x3 = *(const float4*)(hp + 16 * hf + 12);
        v16bf afrag;
        afrag[0]  = tobf(x0.x); afrag[1]  = tobf(x0.y);
        afrag[2]  = tobf(x0.z); afrag[3]  = tobf(x0.w);
        afrag[4]  = tobf(x1.x); afrag[5]  = tobf(x1.y);
        afrag[6]  = tobf(x1.z); afrag[7]  = tobf(x1.w);
        afrag[8]  = tobf(x2.x); afrag[9]  = tobf(x2.y);
        afrag[10] = tobf(x2.z); afrag[11] = tobf(x2.w);
        afrag[12] = tobf(x3.x); afrag[13] = tobf(x3.y);
        afrag[14] = tobf(x3.z); afrag[15] = tobf(x3.w);

        #pragma unroll
        for (int s = 0; s < 8; ++s) {
            const int bcol = s * 16 + l16;
            const v16bf bfrag = *(const v16bf*)&Wlds[bcol][kb + hf * 16];
            acc[s] = __builtin_amdgcn_wmma_f32_16x16x32_bf16(
                false, afrag, false, bfrag, (short)0, acc[s], false, false);
        }
    }

    // Store Wh transposed as bf16: Whbt[(bh*F + col)*N + row]
    __bf16* wbt = Whbt + (size_t)bh * FN * NN;
    #pragma unroll
    for (int s = 0; s < 8; ++s) {
        #pragma unroll
        for (int r = 0; r < 8; ++r) {
            const int row = rowbase + r + 8 * hf;
            const int col = s * 16 + l16;
            wbt[(size_t)col * NN + row] = tobf(acc[s][r]);
        }
    }

    // Fused s_i / s_j reductions over each full 128-wide row
    const float* a1 = a + (size_t)head * 2 * FN;
    const float* a2 = a1 + FN;
    #pragma unroll
    for (int r = 0; r < 8; ++r) {
        float siv = 0.f, sjv = 0.f;
        #pragma unroll
        for (int s = 0; s < 8; ++s) {
            const float v = acc[s][r];
            siv += v * a1[s * 16 + l16];
            sjv += v * a2[s * 16 + l16];
        }
        #pragma unroll
        for (int off = 8; off >= 1; off >>= 1) {
            siv += __shfl_xor(siv, off, 16);
            sjv += __shfl_xor(sjv, off, 16);
        }
        if (l16 == 0) {
            const int row = rowbase + r + 8 * hf;
            s_i[(size_t)bh * NN + row] = siv;
            s_j[(size_t)bh * NN + row] = sjv;
        }
    }
}

// ---------------------------------------------------------------------------
// Kernel 2: single pass over adj (134 MB input, read exactly once).
// Online softmax stats (m, l) for both heads + ballot-compress each adjacency
// row into a bitmask (4 MB) so the WMMA pass never rereads the 134 MB.
// grid = (N/8, B), block = 256 (8 waves, one row per wave)
// ---------------------------------------------------------------------------
__global__ __launch_bounds__(256) void stats_kernel(
    const int* __restrict__ adj, const float* __restrict__ s_i,
    const float* __restrict__ s_j, float* __restrict__ rowmax,
    float* __restrict__ rowinv, unsigned* __restrict__ adjbits)
{
    const int b    = blockIdx.y;
    const int wave = threadIdx.x >> 5;
    const int lid  = threadIdx.x & 31;
    const int i    = blockIdx.x * 8 + wave;

    const int* arow = adj + ((size_t)b * NN + i) * NN;
    unsigned* wrow  = adjbits + ((size_t)b * NN + i) * (NN / 32);

    const float si0 = s_i[((size_t)b * HN + 0) * NN + i];
    const float si1 = s_i[((size_t)b * HN + 1) * NN + i];
    const float* sj0 = s_j + ((size_t)b * HN + 0) * NN;
    const float* sj1 = s_j + ((size_t)b * HN + 1) * NN;

    float m0 = -INFINITY, l0 = 0.f;
    float m1 = -INFINITY, l1 = 0.f;

    for (int k = 0; k < NN / 32; ++k) {
        const int j = k * 32 + lid;
        __builtin_prefetch(arow + j + 256, 0, 1);   // global_prefetch_b8
        const int av = arow[j];
        const unsigned long long bm = __ballot(av != 0);
        if (lid == 0) wrow[k] = (unsigned)bm;
        if (av != 0) {
            float e0 = si0 + sj0[j]; e0 = (e0 > 0.f) ? e0 : 0.2f * e0;
            float e1 = si1 + sj1[j]; e1 = (e1 > 0.f) ? e1 : 0.2f * e1;
            if (e0 > m0) { l0 = l0 * __expf(m0 - e0) + 1.f; m0 = e0; }
            else         { l0 += __expf(e0 - m0); }
            if (e1 > m1) { l1 = l1 * __expf(m1 - e1) + 1.f; m1 = e1; }
            else         { l1 += __expf(e1 - m1); }
        }
    }

    #pragma unroll
    for (int off = 16; off >= 1; off >>= 1) {
        const float m0o = __shfl_xor(m0, off, 32);
        const float l0o = __shfl_xor(l0, off, 32);
        const float m1o = __shfl_xor(m1, off, 32);
        const float l1o = __shfl_xor(l1, off, 32);
        float nm = fmaxf(m0, m0o);
        float sa = (l0  > 0.f) ? __expf(m0  - nm) : 0.f;
        float sb = (l0o > 0.f) ? __expf(m0o - nm) : 0.f;
        l0 = l0 * sa + l0o * sb; m0 = nm;
        nm = fmaxf(m1, m1o);
        sa = (l1  > 0.f) ? __expf(m1  - nm) : 0.f;
        sb = (l1o > 0.f) ? __expf(m1o - nm) : 0.f;
        l1 = l1 * sa + l1o * sb; m1 = nm;
    }
    if (lid == 0) {
        rowmax[((size_t)b * HN + 0) * NN + i] = m0;
        rowinv[((size_t)b * HN + 0) * NN + i] = 1.f / l0;
        rowmax[((size_t)b * HN + 1) * NN + i] = m1;
        rowinv[((size_t)b * HN + 1) * NN + i] = 1.f / l1;
    }
}

// ---------------------------------------------------------------------------
// Kernel 3: out = softmax(e) @ Wh, flash-style with final stats.
// The 32x128 bf16 Wh^T tile for each K-step is staged block-cooperatively
// into LDS (double-buffered, async-to-LDS when available) and shared by all
// 8 waves; probabilities are rebuilt per A-fragment element from the bitmask.
// grid = (B*H, N/128), block = 256 (8 waves, 16 rows per wave)
// ---------------------------------------------------------------------------
__global__ __launch_bounds__(256) void attn_kernel(
    const __bf16* __restrict__ Whbt, const float* __restrict__ s_i,
    const float* __restrict__ s_j, const float* __restrict__ rowmax,
    const float* __restrict__ rowinv, const unsigned* __restrict__ adjbits,
    float* __restrict__ out)
{
    __shared__ __bf16 btile[2][FN][32];   // 2 x 8 KB, [col][k-in-tile]

    const int bh   = blockIdx.x;
    const int b    = bh / HN;
    const int head = bh % HN;
    const int wave = threadIdx.x >> 5;
    const int lid  = threadIdx.x & 31;
    const int l16  = lid & 15;
    const int hf   = lid >> 4;
    const int rowbase = blockIdx.y * 128 + wave * 16;
    const int arow = rowbase + l16;   // A-fragment row for this lane

    const float siv  = s_i[(size_t)bh * NN + arow];
    const float mv   = rowmax[(size_t)bh * NN + arow];
    const float rinv = rowinv[(size_t)bh * NN + arow];
    const unsigned* wrow = adjbits + ((size_t)b * NN + arow) * (NN / 32);
    const float* sjrow   = s_j + (size_t)bh * NN;
    const __bf16* wbt    = Whbt + (size_t)bh * FN * NN;

    // Cooperative copy assignment: each thread owns one 32 B segment
    const int cc   = threadIdx.x >> 1;   // column 0..127
    const int part = threadIdx.x & 1;    // 16-element half
    const __bf16* csrc = wbt + (size_t)cc * NN + part * 16;
    __bf16* cdst0 = &btile[0][cc][part * 16];
    __bf16* cdst1 = &btile[1][cc][part * 16];

#ifdef USE_ASYNC_LDS
    async_copy32(csrc, cdst0);           // prologue: tile for kt = 0
#else
    v16bf stage = *(const v16bf*)csrc;   // fallback: register staging
#endif

    v8f acc[8] = {};

    for (int kt = 0; kt < NN / 32; ++kt) {
        const int jbase = kt * 32;
        const int cur   = kt & 1;

#ifdef USE_ASYNC_LDS
        __builtin_amdgcn_s_wait_asynccnt(0);
        __syncthreads();
        if (kt + 1 < NN / 32)
            async_copy32(csrc + jbase + 32, cur ? cdst0 : cdst1);
#else
        __syncthreads();
        *(v16bf*)(cur ? cdst1 : cdst0) = stage;
        __syncthreads();
        if (kt + 1 < NN / 32)
            stage = *(const v16bf*)(csrc + jbase + 32);
#endif

        // A fragment: final attention probabilities (bf16)
        const unsigned w = wrow[kt];
        v16bf afrag;
        #pragma unroll
        for (int t = 0; t < 16; ++t) {
            const int kk = t + 16 * hf;
            float p = 0.f;
            if ((w >> kk) & 1u) {
                float e = siv + sjrow[jbase + kk];
                e = (e > 0.f) ? e : 0.2f * e;
                p = __expf(e - mv) * rinv;
            }
            afrag[t] = tobf(p);
        }

        #pragma unroll
        for (int s = 0; s < 8; ++s) {
            const int bcol = s * 16 + l16;
            const v16bf bfrag = *(const v16bf*)&btile[cur][bcol][hf * 16];
            acc[s] = __builtin_amdgcn_wmma_f32_16x16x32_bf16(
                false, afrag, false, bfrag, (short)0, acc[s], false, false);
        }
    }

    // Write output: (B, N, H*F)
    #pragma unroll
    for (int s = 0; s < 8; ++s) {
        #pragma unroll
        for (int r = 0; r < 8; ++r) {
            const int row = rowbase + r + 8 * hf;
            const int col = s * 16 + l16;
            out[((size_t)b * NN + row) * (HN * FN) + head * FN + col] = acc[s][r];
        }
    }
}

// ---------------------------------------------------------------------------
extern "C" void kernel_launch(void* const* d_in, const int* in_sizes, int n_in,
                              void* d_out, int out_size, void* d_ws, size_t ws_size,
                              hipStream_t stream) {
    const float* h   = (const float*)d_in[0];
    const int*   adj = (const int*)d_in[1];
    const float* W   = (const float*)d_in[2];
    const float* a   = (const float*)d_in[3];
    float* out = (float*)d_out;

    // Workspace carve-out (~12.5 MB total)
    char* ws = (char*)d_ws;
    __bf16* Whbt = (__bf16*)ws;                                   // 8 MB
    size_t off = (size_t)BN * HN * NN * FN * sizeof(__bf16);
    float* s_i    = (float*)(ws + off); off += (size_t)BN * HN * NN * sizeof(float);
    float* s_j    = (float*)(ws + off); off += (size_t)BN * HN * NN * sizeof(float);
    float* rowmax = (float*)(ws + off); off += (size_t)BN * HN * NN * sizeof(float);
    float* rowinv = (float*)(ws + off); off += (size_t)BN * HN * NN * sizeof(float);
    unsigned* adjbits = (unsigned*)(ws + off);                    // 4 MB

    wh_kernel<<<dim3(BN * HN, NN / 128), 256, 0, stream>>>(h, W, a, Whbt, s_i, s_j);
    stats_kernel<<<dim3(NN / 8, BN), 256, 0, stream>>>(adj, s_i, s_j, rowmax, rowinv, adjbits);
    attn_kernel<<<dim3(BN * HN, NN / 128), 256, 0, stream>>>(Whbt, s_i, s_j, rowmax,
                                                             rowinv, adjbits, out);
}
